// SelfAttentionNarrow_11227044512445
// MI455X (gfx1250) — matrix operand order repaired and checked by reading, not compiled, not run on hardware
//
#include <hip/hip_runtime.h>

// ---------------------------------------------------------------------------
// Self-attention (narrow heads) for gfx1250 (MI455X), wave32 + WMMA +
// async global->LDS staging. B=2, T=2048, EMB=1024, HEADS=16, S=64.
// ---------------------------------------------------------------------------

#define B_ 2
#define T_ 2048
#define EMB_ 1024
#define H_ 16
#define S_ 64

// padded LDS row strides (halfs) -> conflict-free ds_load_b128 fragment reads
#define KPAD 72   // 32 x 72 halfs  (row = 144 B, 16 distinct start banks)
#define VPAD 40   // 64 x 40 halfs  (row =  80 B, 16 distinct start banks)

typedef __attribute__((ext_vector_type(16))) _Float16 v16h;
typedef __attribute__((ext_vector_type(8)))  float    v8f;

union HV {
    v16h v;
    _Float16 h[16];
    uint4 q[2];
};
union H8 { _Float16 h[8]; uint4 q; };

// D = A x B + C,  16x16x32 f16 -> f32
__device__ __forceinline__ v8f wmma16(v16h a, v16h b, v8f c) {
    return __builtin_amdgcn_wmma_f32_16x16x32_f16(
        false, a, false, b, (short)0, c, false, false);
}

// A-fragment (16x32, f16 source): row chunk of 32 halfs at `rowk`.
// element i: K = 2*hi8 + i (contiguous)
__device__ __forceinline__ v16h afrag_f16(const _Float16* rowk, int hi8) {
    HV r;
    r.q[0] = *(const uint4*)(rowk + 2 * hi8);
    r.q[1] = *(const uint4*)(rowk + 2 * hi8 + 8);
    return r.v;
}

// A-fragment from fp32 source (converted to f16).
__device__ __forceinline__ v16h afrag_f32(const float* rowk, int hi8) {
    HV r;
#pragma unroll
    for (int i = 0; i < 8; ++i) r.h[i]     = (_Float16)rowk[2 * hi8 + i];
#pragma unroll
    for (int i = 0; i < 8; ++i) r.h[8 + i] = (_Float16)rowk[2 * hi8 + 8 + i];
    return r.v;
}

// B-fragment (32x16): element i holds K = khalf + i for this lane's column;
// `p` points at column_base + khalf.
__device__ __forceinline__ v16h bfrag_f16(const _Float16* p) {
    HV r;
    r.q[0] = *(const uint4*)(p);
    r.q[1] = *(const uint4*)(p + 8);
    return r.v;
}

// Async global -> LDS copy of 16 bytes per lane (ASYNCcnt-tracked).
__device__ __forceinline__ void async_b128(void* lds, const void* g) {
    asm volatile("global_load_async_to_lds_b128 %0, %1, off"
                 :: "v"((unsigned)(unsigned long long)lds), "v"(g)
                 : "memory");
}
__device__ __forceinline__ void wait_async0() {
    asm volatile("s_wait_asynccnt 0x0" ::: "memory");
}

// ---------------------------------------------------------------------------
// Kernel 0: fp32 -> f16 conversion with scale (weights converted once;
// softmax scale and log2(e) folded into Wq/Wk here).
// ---------------------------------------------------------------------------
__global__ __launch_bounds__(256) void cvt_kernel(
    const float* __restrict__ src, _Float16* __restrict__ dst, float scale) {
    const int i = (blockIdx.x * 256 + threadIdx.x) * 4;
    const float4 f = *(const float4*)(src + i);
    union { _Float16 h[4]; uint2 u; } o;
    o.h[0] = (_Float16)(f.x * scale); o.h[1] = (_Float16)(f.y * scale);
    o.h[2] = (_Float16)(f.z * scale); o.h[3] = (_Float16)(f.w * scale);
    *(uint2*)(dst + i) = o.u;
}

// ---------------------------------------------------------------------------
// Kernel 1: per-head QKV projection.
//   rows r = (b*T + t)*H + h  (H==16 -> one 16-row tile == all heads of a token)
//   Scales pre-folded into Wqh/Wkh. Q,K f16 [b,h,t,s]; V f16 [b,h,s,t].
// ---------------------------------------------------------------------------
__global__ __launch_bounds__(256) void qkv_kernel(
    const float* __restrict__ x,
    const _Float16* __restrict__ Wkh, const _Float16* __restrict__ Wqh,
    const _Float16* __restrict__ Wvh,
    _Float16* __restrict__ Qb, _Float16* __restrict__ Kb,
    _Float16* __restrict__ Vt) {
    const int lane = threadIdx.x & 31;
    const int wave = threadIdx.x >> 5;
    const int row0 = (blockIdx.x * 8 + wave) * 16;
    const int hi    = (lane >> 4) & 1;
    const int hi8   = hi * 8;
    const int nl    = lane & 15;
    const int khalf = hi * 16;

    const float* xrow = x + (size_t)(row0 + nl) * 64;
    v16h a0 = afrag_f32(xrow, hi8);
    v16h a1 = afrag_f32(xrow + 32, hi8);

#pragma unroll
    for (int nb = 0; nb < 4; ++nb) {
        const int col = nb * 16 + nl;
        v8f ck = {}, cq = {}, cv = {};
        ck = wmma16(a0, bfrag_f16(Wkh + (size_t)col * 64 + khalf), ck);
        ck = wmma16(a1, bfrag_f16(Wkh + (size_t)col * 64 + 32 + khalf), ck);
        cq = wmma16(a0, bfrag_f16(Wqh + (size_t)col * 64 + khalf), cq);
        cq = wmma16(a1, bfrag_f16(Wqh + (size_t)col * 64 + 32 + khalf), cq);
        cv = wmma16(a0, bfrag_f16(Wvh + (size_t)col * 64 + khalf), cv);
        cv = wmma16(a1, bfrag_f16(Wvh + (size_t)col * 64 + 32 + khalf), cv);

#pragma unroll
        for (int j = 0; j < 8; ++j) {
            const int r     = row0 + j + hi8;
            const int h     = r & 15;
            const int token = r >> 4;
            const int b     = token >> 11;
            const int t     = token & 2047;
            const int bh    = (b << 4) | h;
            const size_t kq = ((size_t)bh * T_ + t) * S_ + col;
            Kb[kq] = (_Float16)ck[j];
            Qb[kq] = (_Float16)cq[j];
            Vt[((size_t)bh * S_ + col) * T_ + t] = (_Float16)cv[j];
        }
    }
}

// ---------------------------------------------------------------------------
// Kernel 2: flash attention, transposed-score formulation.
//   Block = 8 waves = 128 queries of one (b,h); K/V^T tiles (32 keys) are
//   double-buffered in (bank-conflict-padded) LDS via
//   GLOBAL_LOAD_ASYNC_TO_LDS_B128 and shared by all waves. Scores computed as
//   S^T = K @ Q^T so each lane owns ONE query column; PV as O^T = V^T @ P^T.
// ---------------------------------------------------------------------------
__global__ __launch_bounds__(256) void attn_kernel(
    const _Float16* __restrict__ Qb, const _Float16* __restrict__ Kb,
    const _Float16* __restrict__ Vt, _Float16* __restrict__ Ob) {
    __shared__ _Float16 sK[2][32 * KPAD];   // [key][s]  (padded rows)
    __shared__ _Float16 sV[2][64 * VPAD];   // [d][key]  (padded rows)

    const int tid  = threadIdx.x;
    const int lane = tid & 31;
    const int wave = tid >> 5;

    const int bh = blockIdx.x >> 4;
    const int qt = blockIdx.x & 15;
    const int q0 = qt * 128 + wave * 16;
    const int hi    = (lane >> 4) & 1;
    const int hi8   = hi * 8;
    const int nl    = lane & 15;
    const int khalf = hi * 16;

    const _Float16* Qh = Qb + (size_t)bh * T_ * S_;
    const _Float16* Kh = Kb + (size_t)bh * T_ * S_;
    const _Float16* Vh = Vt + (size_t)bh * S_ * T_;

    // Q^T B-fragments: column q = q0+nl, contraction = s
    const _Float16* qrow = Qh + (size_t)(q0 + nl) * S_;
    const v16h bq0 = bfrag_f16(qrow + khalf);
    const v16h bq1 = bfrag_f16(qrow + 32 + khalf);

    // staging addresses for this thread (one 16 B chunk each in K and V tiles)
    const int kr = tid >> 3;            // 0..31 (key row)
    const int kc = (tid & 7) * 8;       // 0..56 (s offset)
    const int vr = tid >> 2;            // 0..63 (d row)
    const int vc = (tid & 3) * 8;       // 0,8,16,24 (key offset)

    v8f ot0 = {}, ot1 = {}, ot2 = {}, ot3 = {};
    float mq = -1.0e30f, lq = 0.0f;

    // prologue: async-stage keys [0,32) into buffer 0
    async_b128(sK[0] + kr * KPAD + kc, Kh + (size_t)kr * 64 + kc);
    async_b128(sV[0] + vr * VPAD + vc, Vh + (size_t)vr * T_ + vc);

    for (int kb = 0; kb < T_ / 32; ++kb) {
        wait_async0();      // our async LDS writes are done
        __syncthreads();    // everyone's are done; prev buffer fully consumed
        const int cur = kb & 1;
        const int nxt = cur ^ 1;
        if (kb + 1 < T_ / 32) {
            const int kn = (kb + 1) * 32;
            async_b128(sK[nxt] + kr * KPAD + kc,
                       Kh + (size_t)(kn + kr) * 64 + kc);
            async_b128(sV[nxt] + vr * VPAD + vc,
                       Vh + (size_t)vr * T_ + kn + vc);
        }
        const _Float16* lk = sK[cur];
        const _Float16* lv = sV[cur];

        // ---- S^T tile: rows = 32 keys, cols = 16 queries ----
        v8f ct0 = {}, ct1 = {};
        ct0 = wmma16(afrag_f16(lk + (size_t)nl * KPAD, hi8),        bq0, ct0);
        ct0 = wmma16(afrag_f16(lk + (size_t)nl * KPAD + 32, hi8),   bq1, ct0);
        ct1 = wmma16(afrag_f16(lk + (size_t)(16 + nl) * KPAD, hi8),      bq0, ct1);
        ct1 = wmma16(afrag_f16(lk + (size_t)(16 + nl) * KPAD + 32, hi8), bq1, ct1);

        // ---- online softmax (base-2; per-lane query column) ----
        float vmax = ct0[0];
#pragma unroll
        for (int j = 1; j < 8; ++j) vmax = fmaxf(vmax, ct0[j]);
#pragma unroll
        for (int j = 0; j < 8; ++j) vmax = fmaxf(vmax, ct1[j]);
        vmax = fmaxf(vmax, __shfl_xor(vmax, 16, 32));
        const float nm = fmaxf(mq, vmax);
        const float alpha = __builtin_amdgcn_exp2f(mq - nm);
        mq = nm;

        float p0[8], p1[8], ps = 0.0f;
#pragma unroll
        for (int j = 0; j < 8; ++j) {
            p0[j] = __builtin_amdgcn_exp2f(ct0[j] - nm);
            p1[j] = __builtin_amdgcn_exp2f(ct1[j] - nm);
            ps += p0[j] + p1[j];
        }
        ps += __shfl_xor(ps, 16, 32);
        lq = lq * alpha + ps;

#pragma unroll
        for (int j = 0; j < 8; ++j) {
            ot0[j] *= alpha; ot1[j] *= alpha;
            ot2[j] *= alpha; ot3[j] *= alpha;
        }

        // ---- build P^T B-fragment (K = 32 keys) via one cross-half swap ----
        HV pb;
#pragma unroll
        for (int j = 0; j < 8; ++j) {
            const float send = hi ? p0[j] : p1[j];
            const float recv = __shfl_xor(send, 16, 32);
            pb.h[j]     = (_Float16)(hi ? recv : p0[j]);  // K = khalf + j
            pb.h[8 + j] = (_Float16)(hi ? p1[j] : recv);  // K = khalf + 8 + j
        }

        // ---- O^T += V^T @ P^T ----
        ot0 = wmma16(afrag_f16(lv + (size_t)(0  + nl) * VPAD, hi8), pb.v, ot0);
        ot1 = wmma16(afrag_f16(lv + (size_t)(16 + nl) * VPAD, hi8), pb.v, ot1);
        ot2 = wmma16(afrag_f16(lv + (size_t)(32 + nl) * VPAD, hi8), pb.v, ot2);
        ot3 = wmma16(afrag_f16(lv + (size_t)(48 + nl) * VPAD, hi8), pb.v, ot3);
    }

    // ---- finalize: O^T[d][q] -> Ob[b, t=q, h*64+d]; 8 consecutive d per
    //      (dt,j)-group -> packed 16 B stores ----
    const int h = bh & 15, b = bh >> 4;
    const float inv = 1.0f / lq;
    _Float16* op = Ob + ((size_t)b * T_ + (q0 + nl)) * EMB_ + h * S_ + hi8;
    H8 s0, s1, s2, s3;
#pragma unroll
    for (int j = 0; j < 8; ++j) {
        s0.h[j] = (_Float16)(ot0[j] * inv);
        s1.h[j] = (_Float16)(ot1[j] * inv);
        s2.h[j] = (_Float16)(ot2[j] * inv);
        s3.h[j] = (_Float16)(ot3[j] * inv);
    }
    *(uint4*)(op + 0)  = s0.q;
    *(uint4*)(op + 16) = s1.q;
    *(uint4*)(op + 32) = s2.q;
    *(uint4*)(op + 48) = s3.q;
}

// ---------------------------------------------------------------------------
// Kernel 3: output projection Y = O @ Wu^T + bu  (4096 x 1024 x 1024).
// Wave tile = 64 rows x 64 cols (B-fragment reused across 4 row-fragments).
// ---------------------------------------------------------------------------
__global__ __launch_bounds__(256) void outproj_kernel(
    const _Float16* __restrict__ Ob, const _Float16* __restrict__ Wuh,
    const float* __restrict__ bu, float* __restrict__ Y) {
    const int lane = threadIdx.x & 31;
    const int wave = threadIdx.x >> 5;
    const int tile = blockIdx.x * 8 + wave;   // 1024 tiles
    const int r0   = (tile >> 4) * 64;        // 64 row tiles
    const int n0   = (tile & 15) * 64;        // 16 col tiles
    const int hi    = (lane >> 4) & 1;
    const int hi8   = hi * 8;
    const int nl    = lane & 15;
    const int khalf = hi * 16;

    v8f c[4][4];
#pragma unroll
    for (int nt = 0; nt < 4; ++nt) {
        const float bias = bu[n0 + nt * 16 + nl];
#pragma unroll
        for (int rt = 0; rt < 4; ++rt)
#pragma unroll
            for (int j = 0; j < 8; ++j) c[rt][nt][j] = bias;
    }

    for (int kk = 0; kk < EMB_ / 32; ++kk) {
        v16h a[4];
#pragma unroll
        for (int rt = 0; rt < 4; ++rt)
            a[rt] = afrag_f16(Ob + (size_t)(r0 + rt * 16 + nl) * EMB_ + kk * 32, hi8);
#pragma unroll
        for (int nt = 0; nt < 4; ++nt) {
            const v16h bw =
                bfrag_f16(Wuh + (size_t)(n0 + nt * 16 + nl) * EMB_ + kk * 32 + khalf);
#pragma unroll
            for (int rt = 0; rt < 4; ++rt) c[rt][nt] = wmma16(a[rt], bw, c[rt][nt]);
        }
    }

#pragma unroll
    for (int rt = 0; rt < 4; ++rt)
#pragma unroll
        for (int nt = 0; nt < 4; ++nt)
#pragma unroll
            for (int j = 0; j < 8; ++j)
                Y[(size_t)(r0 + rt * 16 + j + hi8) * EMB_ + n0 + nt * 16 + nl] =
                    c[rt][nt][j];
}

// ---------------------------------------------------------------------------
extern "C" void kernel_launch(void* const* d_in, const int* in_sizes, int n_in,
                              void* d_out, int out_size, void* d_ws, size_t ws_size,
                              hipStream_t stream) {
    const float* x  = (const float*)d_in[0];
    const float* Wk = (const float*)d_in[1];
    const float* Wq = (const float*)d_in[2];
    const float* Wv = (const float*)d_in[3];
    const float* Wu = (const float*)d_in[4];
    const float* bu = (const float*)d_in[5];
    float* out = (float*)d_out;

    const size_t HSZ = (size_t)B_ * H_ * T_ * S_;  // 4,194,304 halfs per buffer
    _Float16* Qb  = (_Float16*)d_ws;
    _Float16* Kb  = Qb + HSZ;
    _Float16* Vt  = Kb + HSZ;
    _Float16* Ob  = Vt + HSZ;                  // [b, t, emb] f16
    _Float16* Wuh = Ob + HSZ;                  // 1M halfs
    _Float16* Wkh = Wuh + (size_t)EMB_ * EMB_; // 4096 halfs each
    _Float16* Wqh = Wkh + S_ * S_;
    _Float16* Wvh = Wqh + S_ * S_;             // total ws ~34 MB

    const float kscale = 0.17677669529663687f;          // 1024^-0.25
    const float qscale = kscale * 1.4426950408889634f;  // * log2(e)

    cvt_kernel<<<1024, 256, 0, stream>>>(Wu, Wuh, 1.0f);
    cvt_kernel<<<4, 256, 0, stream>>>(Wk, Wkh, kscale);
    cvt_kernel<<<4, 256, 0, stream>>>(Wq, Wqh, qscale);
    cvt_kernel<<<4, 256, 0, stream>>>(Wv, Wvh, 1.0f);
    qkv_kernel<<<512, 256, 0, stream>>>(x, Wkh, Wqh, Wvh, Qb, Kb, Vt);
    attn_kernel<<<512, 256, 0, stream>>>(Qb, Kb, Vt, Ob);
    outproj_kernel<<<128, 256, 0, stream>>>(Ob, Wuh, bu, out);
}
